// GraphAttention_2894807958201
// MI455X (gfx1250) — compile-verified, matrix operand order checked by reading, not run on hardware
//
#include <hip/hip_runtime.h>

// ---------------------------------------------------------------------------
// Graph attention on gfx1250 (MI455X): bf16 WMMA everywhere, f32 accumulate.
// B=64, N=512, D_IN=D_OUT=256, H=4, HD=64.
// Native __bf16 casts so f32->bf16 lowers to v_cvt[_pk]_bf16_f32 instead of
// a 4-op software round chain.
// ---------------------------------------------------------------------------

typedef __attribute__((ext_vector_type(16))) __bf16 v16bf;
typedef __attribute__((ext_vector_type(8)))  float  v8f;

#define BB   64
#define BN   512
#define DIN  256
#define DOUT 256
#define NH   4
#define HD   64

union ABu { v16bf v; __bf16 e[16]; unsigned u[8]; };

// ---------------------------------------------------------------------------
// Kernel 1: QKV projection.  out = x @ W^T + b, f32 -> bf16.
//   grid (2048, 4, 3), block 128 (4 waves).  Wave computes one 16x16 tile.
//   mat 0 -> Q [b,h,n,hd], mat 1 -> K [b,h,n,hd], mat 2 -> V [b,h,hd,n].
// ---------------------------------------------------------------------------
__global__ __launch_bounds__(128)
void k_qkv(const float* __restrict__ x,
           const float* __restrict__ wq, const float* __restrict__ bq,
           const float* __restrict__ wk, const float* __restrict__ bk,
           const float* __restrict__ wv, const float* __restrict__ bv,
           __bf16* __restrict__ qws,
           __bf16* __restrict__ kws,
           __bf16* __restrict__ vws)
{
    const int lane  = threadIdx.x & 31;
    const int wave  = threadIdx.x >> 5;
    const int mtile = blockIdx.x;              // 0..2047 (rows of B*N)
    const int ntile = blockIdx.y * 4 + wave;   // 0..15   (cols of D_OUT)
    const int mat   = blockIdx.z;              // 0=Q 1=K 2=V

    const float* w    = (mat == 0) ? wq : (mat == 1) ? wk : wv;
    const float* bias = (mat == 0) ? bq : (mat == 1) ? bk : bv;

    const int ln   = lane & 15;
    const int kb   = (lane >> 4) * 8;
    const int arow = mtile * 16 + ln;          // x row (lane's A row)
    const int bcol = ntile * 16 + ln;          // W row == output column

    v8f acc = {};
    #pragma unroll
    for (int kk = 0; kk < 8; ++kk) {           // K = 256 in steps of 32
        ABu A, Bm;
        #pragma unroll
        for (int j = 0; j < 8; ++j) {
            const int k = kk * 32 + (j >> 2) * 16 + kb + 2 * (j & 3);
            A.e[2*j]    = (__bf16)x[arow * DIN + k];
            A.e[2*j+1]  = (__bf16)x[arow * DIN + k + 1];
            Bm.e[2*j]   = (__bf16)w[bcol * DIN + k];
            Bm.e[2*j+1] = (__bf16)w[bcol * DIN + k + 1];
        }
        acc = __builtin_amdgcn_wmma_f32_16x16x32_bf16(
                  false, A.v, false, Bm.v, (short)0, acc, false, false);
    }

    const float badd = bias[bcol];
    const int h  = bcol >> 6;
    const int hd = bcol & 63;
    #pragma unroll
    for (int r = 0; r < 8; ++r) {
        const int row = mtile * 16 + r + 8 * (lane >> 4);
        const int b   = row >> 9;              // /512
        const int n   = row & 511;
        const __bf16 ob = (__bf16)(acc[r] + badd);
        if (mat == 0)      qws[(((size_t)b * NH + h) * BN + n) * HD + hd] = ob;
        else if (mat == 1) kws[(((size_t)b * NH + h) * BN + n) * HD + hd] = ob;
        else               vws[(((size_t)b * NH + h) * HD + hd) * BN + n] = ob;
    }
}

// ---------------------------------------------------------------------------
// Kernel 2: masked attention, one wave per (b, h, 16-query tile).
//   S = (Q K^T)/8 masked by adj -> LDS; row softmax; O = P V via WMMA.
//   grid 8192, block 32.  LDS: 32KB f32 scores + 16KB bf16 P.
// ---------------------------------------------------------------------------
__global__ __launch_bounds__(32)
void k_attn(const __bf16* __restrict__ qws,
            const __bf16* __restrict__ kws,
            const __bf16* __restrict__ vws,
            const int* __restrict__ adj,
            __bf16* __restrict__ aO)
{
    __shared__ float  sS[16 * 512];
    __shared__ __bf16 sP[16 * 512];

    const int lane = threadIdx.x;
    const int idx  = blockIdx.x;
    const int bh   = idx >> 5;                 // b*4 + h
    const int qt   = idx & 31;
    const int q0   = qt * 16;
    const int ln   = lane & 15;
    const int kb   = (lane >> 4) * 8;

    const __bf16* Qb = qws + (size_t)bh * BN * HD;
    const __bf16* Kb = kws + (size_t)bh * BN * HD;
    const __bf16* Vb = vws + (size_t)bh * HD * BN;

    // Q tile (16 x 64) as two A fragments (K chunks of 32 over HD).
    ABu aQ[2];
    #pragma unroll
    for (int c = 0; c < 2; ++c)
        #pragma unroll
        for (int j = 0; j < 8; ++j) {
            const int hd = c * 32 + (j >> 2) * 16 + kb + 2 * (j & 3);
            aQ[c].u[j] = *(const unsigned*)(Qb + (q0 + ln) * HD + hd);
        }

    // ---- scores: 32 key tiles --------------------------------------------
    for (int m = 0; m < 32; ++m) {
        if (m < 31)   // hint the next K tile toward the caches
            __builtin_prefetch(Kb + ((m + 1) * 16 + ln) * HD, 0, 0);
        ABu bK[2];
        #pragma unroll
        for (int c = 0; c < 2; ++c)
            #pragma unroll
            for (int j = 0; j < 8; ++j) {
                const int hd = c * 32 + (j >> 2) * 16 + kb + 2 * (j & 3);
                bK[c].u[j] = *(const unsigned*)(Kb + (m * 16 + ln) * HD + hd);
            }
        v8f s = {};
        s = __builtin_amdgcn_wmma_f32_16x16x32_bf16(
                false, aQ[0].v, false, bK[0].v, (short)0, s, false, false);
        s = __builtin_amdgcn_wmma_f32_16x16x32_bf16(
                false, aQ[1].v, false, bK[1].v, (short)0, s, false, false);

        const int kcol = m * 16 + ln;
        #pragma unroll
        for (int r = 0; r < 8; ++r) {
            const int rl = r + 8 * (lane >> 4);        // local query row
            float val = s[r] * 0.125f;                 // 1/sqrt(64)
            if (adj[(q0 + rl) * BN + kcol] == 0) val = -1.0e30f;
            sS[rl * 512 + kcol] = val;
        }
    }
    __syncthreads();

    // ---- row softmax: 2 lanes per row, 16-lane xor reduction --------------
    {
        const int row = lane & 15;
        const int c0  = lane >> 4;
        float mx = -3.0e38f;
        for (int c = c0; c < 512; c += 2) mx = fmaxf(mx, sS[row * 512 + c]);
        mx = fmaxf(mx, __shfl_xor(mx, 16, 32));
        float sum = 0.f;
        for (int c = c0; c < 512; c += 2) sum += __expf(sS[row * 512 + c] - mx);
        sum += __shfl_xor(sum, 16, 32);
        const float inv = 1.0f / sum;                  // >= 1 term (self-loop)
        for (int c = c0; c < 512; c += 2)
            sP[row * 512 + c] = (__bf16)(__expf(sS[row * 512 + c] - mx) * inv);
    }
    __syncthreads();

    // ---- O = P (16x512) @ V (512x64) -------------------------------------
    const int b = bh >> 2;
    const int h = bh & 3;
    #pragma unroll
    for (int t = 0; t < 4; ++t) {                      // 4 hd tiles of 16
        v8f o = {};
        for (int kc = 0; kc < 16; ++kc) {              // key chunks of 32
            ABu A, Bm;
            #pragma unroll
            for (int j = 0; j < 8; ++j) {
                const int key = kc * 32 + (j >> 2) * 16 + kb + 2 * (j & 3);
                A.u[j]  = *(const unsigned*)(sP + ln * 512 + key);
                Bm.u[j] = *(const unsigned*)(Vb + (t * 16 + ln) * BN + key);
            }
            o = __builtin_amdgcn_wmma_f32_16x16x32_bf16(
                    false, A.v, false, Bm.v, (short)0, o, false, false);
        }
        #pragma unroll
        for (int r = 0; r < 8; ++r) {
            const int rl = r + 8 * (lane >> 4);
            const int n  = q0 + rl;
            aO[((size_t)(b * BN + n)) * DOUT + h * HD + t * 16 + ln] =
                (__bf16)o[r];
        }
    }
}

// ---------------------------------------------------------------------------
// Kernel 3: output projection  out = attnO @ wo^T + bo  (bf16 A, f32 out).
//   grid (2048, 4), block 128.
// ---------------------------------------------------------------------------
__global__ __launch_bounds__(128)
void k_outproj(const __bf16* __restrict__ aO,
               const float* __restrict__ wo, const float* __restrict__ bo,
               float* __restrict__ out)
{
    const int lane  = threadIdx.x & 31;
    const int wave  = threadIdx.x >> 5;
    const int mtile = blockIdx.x;
    const int ntile = blockIdx.y * 4 + wave;
    const int ln    = lane & 15;
    const int kb    = (lane >> 4) * 8;
    const int arow  = mtile * 16 + ln;
    const int bcol  = ntile * 16 + ln;

    v8f acc = {};
    #pragma unroll
    for (int kk = 0; kk < 8; ++kk) {
        ABu A, Bm;
        #pragma unroll
        for (int j = 0; j < 8; ++j) {
            const int k = kk * 32 + (j >> 2) * 16 + kb + 2 * (j & 3);
            A.u[j]      = *(const unsigned*)(aO + (size_t)arow * DOUT + k);
            Bm.e[2*j]   = (__bf16)wo[bcol * DOUT + k];
            Bm.e[2*j+1] = (__bf16)wo[bcol * DOUT + k + 1];
        }
        acc = __builtin_amdgcn_wmma_f32_16x16x32_bf16(
                  false, A.v, false, Bm.v, (short)0, acc, false, false);
    }
    const float badd = bo[bcol];
    #pragma unroll
    for (int r = 0; r < 8; ++r) {
        const int row = mtile * 16 + r + 8 * (lane >> 4);
        out[(size_t)row * DOUT + bcol] = acc[r] + badd;
    }
}

// ---------------------------------------------------------------------------
extern "C" void kernel_launch(void* const* d_in, const int* in_sizes, int n_in,
                              void* d_out, int out_size, void* d_ws, size_t ws_size,
                              hipStream_t stream)
{
    const float* x  = (const float*)d_in[0];
    const int*   adj= (const int*)  d_in[1];
    const float* wq = (const float*)d_in[2];
    const float* bq = (const float*)d_in[3];
    const float* wk = (const float*)d_in[4];
    const float* bk = (const float*)d_in[5];
    const float* wv = (const float*)d_in[6];
    const float* bv = (const float*)d_in[7];
    const float* wo = (const float*)d_in[8];
    const float* bo = (const float*)d_in[9];
    float* out = (float*)d_out;

    const size_t elems = (size_t)BB * NH * BN * HD;      // 8,388,608 bf16 each
    __bf16* qws = (__bf16*)d_ws;
    __bf16* kws = qws + elems;
    __bf16* vws = kws + elems;
    __bf16* aOw = vws + elems;                           // ~64 MB total ws

    dim3 g1(2048, 4, 3), b1(128);
    k_qkv<<<g1, b1, 0, stream>>>(x, wq, bq, wk, bk, wv, bv, qws, kws, vws);

    dim3 g2(BB * NH * 32), b2(32);
    k_attn<<<g2, b2, 0, stream>>>(qws, kws, vws, adj, aOw);

    dim3 g3(2048, 4), b3(128);
    k_outproj<<<g3, b3, 0, stream>>>(aOw, wo, bo, out);
}
